// CILRS_49031346651226
// MI455X (gfx1250) — matrix operand order repaired and checked by reading, compiled, not run on hardware
//
#include <hip/hip_runtime.h>
#include <hip/hip_bf16.h>

typedef __attribute__((ext_vector_type(16))) _Float16 v16h;
typedef __attribute__((ext_vector_type(8)))  float    v8f;

#define CDIV(a, b) (((a) + (b) - 1) / (b))

// ---------------------------------------------------------------------------
// Layout conversion kernels
// ---------------------------------------------------------------------------

// NCHW f32 -> NHWC f16
__global__ void nchw_to_nhwc_f16(const float* __restrict__ x, _Float16* __restrict__ y,
                                 int N, int C, int H, int W) {
  size_t idx = (size_t)blockIdx.x * blockDim.x + threadIdx.x;
  size_t total = (size_t)N * C * H * W;
  if (idx >= total) return;
  int c = (int)(idx % C);
  size_t t = idx / C;
  int w = (int)(t % W); t /= W;
  int h = (int)(t % H);
  int n = (int)(t / H);
  y[idx] = (_Float16)x[(((size_t)n * C + c) * H + h) * W + w];
}

// OIHW f32 -> WMMA B-fragment order f16.
// Layout: [kt][ct][lane][e] with kt = k-dim tile of 32 (row = kt*32 + (lane>>4)*8 + (e&7) + (e>>3)*16),
// ct = 16-col tile (col = ct*16 + (lane&15)). Rows >= RSC are zero padding.
// Logical GEMM row = (r*S + s)*C + c, col = output channel k.
__global__ void pack_weight_frag(const float* __restrict__ w, _Float16* __restrict__ out,
                                 int K, int C, int R, int S, int RSC, int RSCpad) {
  size_t idx = (size_t)blockIdx.x * blockDim.x + threadIdx.x;
  size_t total = (size_t)RSCpad * K;
  if (idx >= total) return;
  int e    = (int)(idx & 15);
  int lane = (int)((idx >> 4) & 31);
  size_t t = idx >> 9;
  int ctiles = K >> 4;
  int ct = (int)(t % ctiles);
  int kt = (int)(t / ctiles);
  int col = (ct << 4) + (lane & 15);
  int row = (kt << 5) + ((lane >> 4) << 3) + (e & 7) + ((e >> 3) << 4);
  float val = 0.0f;
  if (row < RSC) {
    int c  = row % C;
    int rs = row / C;
    int s  = rs % S;
    int r  = rs / S;
    val = w[(((size_t)col * C + c) * R + r) * S + s];
  }
  out[idx] = (_Float16)val;
}

// ---------------------------------------------------------------------------
// Fused implicit-GEMM conv + BN + (residual) + (ReLU), NHWC f16, WMMA f16->f32
// Block = 128 threads = 4 waves arranged 2(M) x 2(N); each wave computes a
// 32(M) x 32(N) tile with 2x2 accumulators sharing 2 A-frags and 2 B-frags
// (8 b128 loads per 4 back-to-back independent WMMAs).
// Grid: (M/64, K/64), M = N*Ho*Wo.
// ---------------------------------------------------------------------------

struct ConvParams {
  const _Float16* x;     // NHWC input
  const _Float16* w;     // B-fragment packed weights
  const float* bng;
  const float* bnb;
  const float* bnm;
  const float* bnv;
  const _Float16* res;   // NHWC residual (may be nullptr)
  _Float16* y;           // NHWC output
  int N, H, W, C;
  int K;
  int stride, pad;
  int Ho, Wo;
  int relu;
};

template <int R, int S, int C>
__global__ __launch_bounds__(128) void conv_bn_wmma(ConvParams p) {
  constexpr int RSC    = R * S * C;
  constexpr int RSCpad = (RSC + 31) & ~31;
  constexpr int KTILES = RSCpad / 32;

  const int lane  = threadIdx.x & 31;
  const int wv    = threadIdx.x >> 5;
  const int mwave = wv >> 1;           // 0/1 : which 32-row half of 64-row tile
  const int nwave = wv & 1;            // 0/1 : which 32-col half of 64-col tile
  const int l15   = lane & 15;
  const int kgrp  = lane >> 4;

  const int m0 = blockIdx.x * 64 + (mwave << 5);   // wave rows m0..m0+31
  const int n0 = blockIdx.y * 64 + (nwave << 5);   // wave cols n0..n0+31

  // Decode A rows for both 16-row sub-tiles (rows duplicated across half-waves)
  int hbase[2], wbase[2];
  const _Float16* __restrict__ xn[2];
  {
#pragma unroll
    for (int mt = 0; mt < 2; ++mt) {
      const int mrow = m0 + (mt << 4) + l15;
      const int wo   = mrow % p.Wo;
      int tt         = mrow / p.Wo;
      const int ho   = tt % p.Ho;
      const int nb   = tt / p.Ho;
      hbase[mt] = ho * p.stride - p.pad;
      wbase[mt] = wo * p.stride - p.pad;
      xn[mt]    = p.x + (size_t)nb * p.H * p.W * C;
    }
  }

  // B fragment addressing: packed as [kt][ct][lane][16]; this wave's first ct:
  const int    ctiles = p.K >> 4;
  const int    ctb    = n0 >> 4;
  const size_t bstep  = (size_t)ctiles << 9;       // halfs per kt tile
  const _Float16* __restrict__ bb = p.w + (((size_t)ctb << 5) + lane) * 16;

  union Frag { uint4 u[2]; v16h v; };
  v8f acc[2][2] = {};

  for (int kt = 0; kt < KTILES; ++kt) {
    const int kb = kt << 5;

    // ---- A fragments (2 row sub-tiles) ----
    Frag ua[2];
    if constexpr (C % 32 == 0) {
      // One filter tap per 32-chunk: two contiguous 16B channel loads per frag.
      const int rs = kb / C;                  // constexpr divisor -> shift
      const int sx = rs % S;
      const int rx = rs / S;
      const int cb = (kb % C) + (kgrp << 3);
#pragma unroll
      for (int mt = 0; mt < 2; ++mt) {
        const int hi = hbase[mt] + rx;
        const int wi = wbase[mt] + sx;
        if (hi >= 0 && hi < p.H && wi >= 0 && wi < p.W) {
          const _Float16* ap = xn[mt] + ((size_t)hi * p.W + wi) * C + cb;
          ua[mt].u[0] = *reinterpret_cast<const uint4*>(ap);       // K kb+kg*8..
          ua[mt].u[1] = *reinterpret_cast<const uint4*>(ap + 16);  // K kb+16+kg*8..
        } else {
          ua[mt].u[0] = uint4{0u, 0u, 0u, 0u};
          ua[mt].u[1] = uint4{0u, 0u, 0u, 0u};
        }
      }
    } else {
      // Generic scalar gather (stem only, C=3): constexpr divisors.
#pragma unroll
      for (int mt = 0; mt < 2; ++mt) {
        v16h a;
#pragma unroll
        for (int e = 0; e < 16; ++e) {
          const int kk = kb + (kgrp << 3) + (e & 7) + ((e >> 3) << 4);
          _Float16 val = (_Float16)0.0f;
          if (kk < RSC) {
            const int c  = kk % C;
            const int rs = kk / C;
            const int sx = rs % S;
            const int rx = rs / S;
            const int hi = hbase[mt] + rx;
            const int wi = wbase[mt] + sx;
            if (hi >= 0 && hi < p.H && wi >= 0 && wi < p.W)
              val = xn[mt][((size_t)hi * p.W + wi) * C + c];
          }
          a[e] = val;
        }
        ua[mt].v = a;
      }
    }

    // ---- B fragments (2 col sub-tiles) + prefetch next k-tile ----
    const _Float16* bp = bb + (size_t)kt * bstep;
    Frag ub[2];
#pragma unroll
    for (int nt = 0; nt < 2; ++nt) {
      const _Float16* bq = bp + (nt << 9);         // +nt*32lanes*16
      ub[nt].u[0] = *reinterpret_cast<const uint4*>(bq);
      ub[nt].u[1] = *reinterpret_cast<const uint4*>(bq + 8);
    }
    if (kt + 1 < KTILES) {
      __builtin_prefetch(bp + bstep, 0, 3);        // global_prefetch_b8
      __builtin_prefetch(bp + bstep + 512, 0, 3);
    }

    // ---- 4 independent WMMAs (no D->A/B hazard, no NOPs needed) ----
#pragma unroll
    for (int mt = 0; mt < 2; ++mt)
#pragma unroll
      for (int nt = 0; nt < 2; ++nt)
        acc[mt][nt] = __builtin_amdgcn_wmma_f32_16x16x32_f16(
            false, ua[mt].v, false, ub[nt].v, (short)0, acc[mt][nt], false, false);
  }

  // ---- Epilogue: fused BN (+ residual) (+ ReLU), NHWC f16 store ----
  const bool hasres = (p.res != nullptr);
  const bool dorelu = (p.relu != 0);

  float scl[2], bias[2];
  int   col[2];
#pragma unroll
  for (int nt = 0; nt < 2; ++nt) {
    col[nt]  = n0 + (nt << 4) + l15;
    scl[nt]  = rsqrtf(p.bnv[col[nt]] + 1e-5f) * p.bng[col[nt]];
    bias[nt] = p.bnb[col[nt]] - p.bnm[col[nt]] * scl[nt];
  }

#pragma unroll
  for (int mt = 0; mt < 2; ++mt) {
    // Incremental M-row decode (one div/mod pair per sub-tile).
    const int mstart = m0 + (mt << 4) + (kgrp << 3);
    int wo2 = mstart % p.Wo;
    int td  = mstart / p.Wo;
    int ho2 = td % p.Ho;
    int nb2 = td / p.Ho;
#pragma unroll
    for (int j = 0; j < 8; ++j) {
      const size_t ridx =
          (((size_t)nb2 * p.Ho + ho2) * p.Wo + wo2) * (size_t)p.K;
#pragma unroll
      for (int nt = 0; nt < 2; ++nt) {
        const size_t oidx = ridx + col[nt];
        float v = acc[mt][nt][j] * scl[nt] + bias[nt];
        if (hasres) v += (float)p.res[oidx];
        if (dorelu) v = fmaxf(v, 0.0f);
        p.y[oidx] = (_Float16)v;
      }
      if (++wo2 == p.Wo) { wo2 = 0; if (++ho2 == p.Ho) { ho2 = 0; ++nb2; } }
    }
  }
}

// ---------------------------------------------------------------------------
// Pooling kernels (NHWC f16)
// ---------------------------------------------------------------------------

__global__ void maxpool3x3s2_nhwc(const _Float16* __restrict__ x, _Float16* __restrict__ y,
                                  int N, int H, int W, int C, int Ho, int Wo) {
  size_t idx = (size_t)blockIdx.x * blockDim.x + threadIdx.x;
  size_t total = (size_t)N * Ho * Wo * C;
  if (idx >= total) return;
  int c = (int)(idx & (C - 1));            // C = 64 (power of two)
  size_t t = idx / C;
  int wo = (int)(t % Wo); t /= Wo;
  int ho = (int)(t % Ho);
  int n  = (int)(t / Ho);
  float m = -3.0e38f;
  for (int r = 0; r < 3; ++r) {
    int hi = ho * 2 - 1 + r;
    if (hi < 0 || hi >= H) continue;
    for (int s = 0; s < 3; ++s) {
      int wi = wo * 2 - 1 + s;
      if (wi < 0 || wi >= W) continue;
      float v = (float)x[(((size_t)n * H + hi) * W + wi) * C + c];
      m = fmaxf(m, v);
    }
  }
  y[idx] = (_Float16)m;
}

// Global average pool: NHWC f16 (N, HW, C) -> f32 [N][C]
__global__ void avgpool_nhwc(const _Float16* __restrict__ x, float* __restrict__ y,
                             int N, int HW, int C) {
  int idx = blockIdx.x * blockDim.x + threadIdx.x;
  if (idx >= N * C) return;
  int c = idx & (C - 1);                   // C = 512
  int n = idx / C;
  float s = 0.0f;
  for (int i = 0; i < HW; ++i)
    s += (float)x[((size_t)n * HW + i) * C + c];
  y[idx] = s / (float)HW;
}

// ---------------------------------------------------------------------------
// Head kernels (tiny f32 GEMMs / per-command branches)
// act: 0 = none, 1 = relu, 2 = sigmoid, 3 = tanh
// ---------------------------------------------------------------------------

__device__ __forceinline__ float apply_act(float s, int act) {
  if (act == 1) return fmaxf(s, 0.0f);
  if (act == 2) return 1.0f / (1.0f + __expf(-s));
  if (act == 3) return tanhf(s);
  return s;
}

__global__ void linear_act(const float* __restrict__ x, int ldx,
                           const float* __restrict__ w, const float* __restrict__ b,
                           float* __restrict__ y, int ldy,
                           int M, int Kin, int Nout, int act) {
  int idx = blockIdx.x * blockDim.x + threadIdx.x;
  if (idx >= M * Nout) return;
  int mr = idx / Nout;
  int n  = idx % Nout;
  float s = b[n];
  for (int k = 0; k < Kin; ++k)
    s += x[(size_t)mr * ldx + k] * w[(size_t)k * Nout + n];
  y[(size_t)mr * ldy + n] = apply_act(s, act);
}

// w: [4][Kin][Nout], b: [4][Nout], per-row command gather
__global__ void branch_linear_act(const float* __restrict__ x, int ldx,
                                  const float* __restrict__ w, const float* __restrict__ b,
                                  const int* __restrict__ cmd,
                                  float* __restrict__ y, int ldy,
                                  int M, int Kin, int Nout, int act) {
  int idx = blockIdx.x * blockDim.x + threadIdx.x;
  if (idx >= M * Nout) return;
  int mr = idx / Nout;
  int n  = idx % Nout;
  int c  = cmd[mr];
  const float* wp = w + (size_t)c * Kin * Nout;
  float s = b[(size_t)c * Nout + n];
  for (int k = 0; k < Kin; ++k)
    s += x[(size_t)mr * ldx + k] * wp[(size_t)k * Nout + n];
  y[(size_t)mr * ldy + n] = apply_act(s, act);
}

// ---------------------------------------------------------------------------
// Host side
// ---------------------------------------------------------------------------

namespace {

struct BN { const float *g, *b, *m, *v; };
struct BlockP {
  const float* w1; BN bn1;
  const float* w2; BN bn2;
  const float* dw; BN dbn;
  bool down;
};
struct BrP { const float *w1, *b1, *w2, *b2, *w3, *b3; };

inline ConvParams make_params(const _Float16* x, const _Float16* w16, const BN& bn,
                              const _Float16* res, _Float16* y,
                              int N, int H, int W, int C, int K,
                              int R, int S, int stride, int pad, int relu) {
  ConvParams p;
  p.x = x; p.w = w16;
  p.bng = bn.g; p.bnb = bn.b; p.bnm = bn.m; p.bnv = bn.v;
  p.res = res; p.y = y;
  p.N = N; p.H = H; p.W = W; p.C = C; p.K = K;
  p.stride = stride; p.pad = pad;
  p.Ho = (H + 2 * pad - R) / stride + 1;
  p.Wo = (W + 2 * pad - S) / stride + 1;
  p.relu = relu;
  return p;
}

template <int R, int S, int C>
inline void launch_conv_t(hipStream_t stream, const ConvParams& p) {
  dim3 grid((unsigned)((p.N * p.Ho * p.Wo) / 64), (unsigned)(p.K / 64));
  conv_bn_wmma<R, S, C><<<grid, 128, 0, stream>>>(p);
}

inline void launch_conv(hipStream_t stream,
                        const _Float16* x, const _Float16* w16, const BN& bn,
                        const _Float16* res, _Float16* y,
                        int N, int H, int W, int C, int K,
                        int R, int S, int stride, int pad, int relu) {
  ConvParams p = make_params(x, w16, bn, res, y, N, H, W, C, K, R, S, stride, pad, relu);
  if (R == 7) {
    launch_conv_t<7, 7, 3>(stream, p);
  } else if (R == 3) {
    switch (C) {
      case 64:  launch_conv_t<3, 3, 64>(stream, p);  break;
      case 128: launch_conv_t<3, 3, 128>(stream, p); break;
      case 256: launch_conv_t<3, 3, 256>(stream, p); break;
      default:  launch_conv_t<3, 3, 512>(stream, p); break;
    }
  } else {
    switch (C) {
      case 64:  launch_conv_t<1, 1, 64>(stream, p);  break;
      case 128: launch_conv_t<1, 1, 128>(stream, p); break;
      default:  launch_conv_t<1, 1, 256>(stream, p); break;
    }
  }
}

} // namespace

extern "C" void kernel_launch(void* const* d_in, const int* in_sizes, int n_in,
                              void* d_out, int out_size, void* d_ws, size_t ws_size,
                              hipStream_t stream) {
  (void)in_sizes; (void)n_in; (void)out_size; (void)ws_size;

  // ---- Walk inputs in setup_inputs() dict insertion order --------------------
  int cur = 0;
  auto F = [&]() { return (const float*)d_in[cur++]; };

  const float* img     = F();                       // (64,3,224,224)
  const int*   command = (const int*)d_in[cur++];   // (64,1)
  const float* speed   = F();                       // (64,1)

  const float* stem_w = F();                        // (64,3,7,7)
  BN stem_bn = { F(), F(), F(), F() };              // g,b,m,v

  static const int outcs[4]   = {64, 128, 256, 512};
  static const int strides[4] = {1, 2, 2, 2};
  BlockP blocks[8];
  {
    int inc = 64, bi = 0;
    for (int st = 0; st < 4; ++st) {
      for (int blk = 0; blk < 2; ++blk, ++bi) {
        int s = (blk == 0) ? strides[st] : 1;
        BlockP bp;
        bp.w1 = F(); bp.bn1 = { F(), F(), F(), F() };
        bp.w2 = F(); bp.bn2 = { F(), F(), F(), F() };
        bp.down = (s != 1 || inc != outcs[st]);
        bp.dw = nullptr; bp.dbn = { nullptr, nullptr, nullptr, nullptr };
        if (bp.down) { bp.dw = F(); bp.dbn = { F(), F(), F(), F() }; }
        blocks[bi] = bp;
        inc = outcs[st];
      }
    }
  }
  const float* fc_w = F(); const float* fc_b = F();
  const float *si_w1 = F(), *si_b1 = F(), *si_w2 = F(), *si_b2 = F();
  const float *sp_w1 = F(), *sp_b1 = F(), *sp_w2 = F(), *sp_b2 = F();
  auto readBr = [&]() { BrP p; p.w1=F(); p.b1=F(); p.w2=F(); p.b2=F(); p.w3=F(); p.b3=F(); return p; };
  BrP thr = readBr();
  BrP brk = readBr();
  BrP str = readBr();

  float* out = (float*)d_out;   // [0..191] actions (64x3), [192..255] speed (64x1)

  // ---- Workspace bump allocator ---------------------------------------------
  char* wsp = (char*)d_ws;
  size_t wsoff = 0;
  auto alloc = [&](size_t bytes) -> void* {
    wsoff = (wsoff + 255) & ~(size_t)255;
    void* p = wsp + wsoff;
    wsoff += bytes;
    return p;
  };

  const int NB = 64;
  const size_t ACT_MAX = (size_t)NB * 112 * 112 * 64;  // max NHWC activation elems

  _Float16* imgh = (_Float16*)alloc((size_t)NB * 224 * 224 * 3 * sizeof(_Float16));
  _Float16* bufs[4];
  for (int i = 0; i < 4; ++i) bufs[i] = (_Float16*)alloc(ACT_MAX * sizeof(_Float16));

  // ---- Pack all conv weights f32 OIHW -> f16 B-fragment layout --------------
  auto pack = [&](const float* src, int K, int C, int R, int S) -> _Float16* {
    int RSC = R * S * C;
    int RSCpad = (RSC + 31) & ~31;
    size_t total = (size_t)RSCpad * K;
    _Float16* dst = (_Float16*)alloc(total * sizeof(_Float16));
    pack_weight_frag<<<(unsigned)CDIV(total, 256), 256, 0, stream>>>(
        src, dst, K, C, R, S, RSC, RSCpad);
    return dst;
  };

  _Float16* stem_w16 = pack(stem_w, 64, 3, 7, 7);
  struct BW16 { _Float16 *w1, *w2, *dw; } w16b[8];
  {
    int inc = 64, bi = 0;
    for (int st = 0; st < 4; ++st) {
      for (int blk = 0; blk < 2; ++blk, ++bi) {
        int outc = outcs[st];
        w16b[bi].w1 = pack(blocks[bi].w1, outc, inc, 3, 3);
        w16b[bi].w2 = pack(blocks[bi].w2, outc, outc, 3, 3);
        w16b[bi].dw = blocks[bi].down ? pack(blocks[bi].dw, outc, inc, 1, 1) : nullptr;
        inc = outc;
      }
    }
  }

  // ---- Image NCHW f32 -> NHWC f16 -------------------------------------------
  {
    size_t total = (size_t)NB * 3 * 224 * 224;
    nchw_to_nhwc_f16<<<(unsigned)CDIV(total, 256), 256, 0, stream>>>(
        img, imgh, NB, 3, 224, 224);
  }

  // ---- Stem: conv7x7 s2 p3 + BN + ReLU -> (64,112,112,64), then maxpool -----
  launch_conv(stream, imgh, stem_w16, stem_bn, nullptr, bufs[0],
              NB, 224, 224, 3, 64, 7, 7, 2, 3, /*relu=*/1);
  {
    size_t total = (size_t)NB * 56 * 56 * 64;
    maxpool3x3s2_nhwc<<<(unsigned)CDIV(total, 256), 256, 0, stream>>>(
        bufs[0], bufs[1], NB, 112, 112, 64, 56, 56);
  }

  // ---- Residual stages -------------------------------------------------------
  int xi = 1;           // bufs[1] holds current activation
  int curH = 56, curC = 64;
  {
    int bi = 0;
    for (int st = 0; st < 4; ++st) {
      for (int blk = 0; blk < 2; ++blk, ++bi) {
        const BlockP& bp = blocks[bi];
        int s = (blk == 0) ? strides[st] : 1;
        int outc = outcs[st];
        int Hout = (curH + 2 - 3) / s + 1;

        int j[3], jn = 0;
        for (int q = 0; q < 4; ++q) if (q != xi) j[jn++] = q;
        _Float16* y1  = bufs[j[0]];
        _Float16* y2  = bufs[j[1]];
        _Float16* scb = bufs[j[2]];

        // conv1 + bn1 + relu
        launch_conv(stream, bufs[xi], w16b[bi].w1, bp.bn1, nullptr, y1,
                    NB, curH, curH, curC, outc, 3, 3, s, 1, /*relu=*/1);
        // shortcut
        const _Float16* res;
        if (bp.down) {
          launch_conv(stream, bufs[xi], w16b[bi].dw, bp.dbn, nullptr, scb,
                      NB, curH, curH, curC, outc, 1, 1, s, 0, /*relu=*/0);
          res = scb;
        } else {
          res = bufs[xi];
        }
        // conv2 + bn2 + residual + relu
        launch_conv(stream, y1, w16b[bi].w2, bp.bn2, res, y2,
                    NB, Hout, Hout, outc, outc, 3, 3, 1, 1, /*relu=*/1);

        xi = j[1];
        curH = Hout;
        curC = outc;
      }
    }
  }

  // ---- Heads -----------------------------------------------------------------
  float* feat512 = (float*)alloc((size_t)NB * 512 * sizeof(float));
  float* cond    = (float*)alloc((size_t)NB * 384 * sizeof(float)); // [img256 | sp128]
  float* h128    = (float*)alloc((size_t)NB * 128 * sizeof(float));
  float* hA      = (float*)alloc((size_t)NB * 256 * sizeof(float));
  float* hB      = (float*)alloc((size_t)NB * 256 * sizeof(float));

  avgpool_nhwc<<<(unsigned)CDIV(NB * 512, 256), 256, 0, stream>>>(
      bufs[xi], feat512, NB, 49, 512);

  // fc (512->256) -> cond[:, 0:256], no activation
  linear_act<<<(unsigned)CDIV(NB * 256, 256), 256, 0, stream>>>(
      feat512, 512, fc_w, fc_b, cond, 384, NB, 512, 256, 0);

  // speed_in MLP: 1->128 relu, 128->128 -> cond[:, 256:384]
  linear_act<<<(unsigned)CDIV(NB * 128, 128), 128, 0, stream>>>(
      speed, 1, si_w1, si_b1, h128, 128, NB, 1, 128, 1);
  linear_act<<<(unsigned)CDIV(NB * 128, 128), 128, 0, stream>>>(
      h128, 128, si_w2, si_b2, cond + 256, 384, NB, 128, 128, 0);

  // speed_pred MLP on img features: 256->256 relu, 256->1 -> out[192..255]
  linear_act<<<(unsigned)CDIV(NB * 256, 256), 256, 0, stream>>>(
      cond, 384, sp_w1, sp_b1, hA, 256, NB, 256, 256, 1);
  linear_act<<<1, 64, 0, stream>>>(
      hA, 256, sp_w2, sp_b2, out + (size_t)NB * 3, 1, NB, 256, 1, 0);

  // command branches: 384->256 relu, 256->256 relu, 256->1 act
  auto run_branch = [&](const BrP& bp, int act, float* dst) {
    branch_linear_act<<<(unsigned)CDIV(NB * 256, 256), 256, 0, stream>>>(
        cond, 384, bp.w1, bp.b1, command, hA, 256, NB, 384, 256, 1);
    branch_linear_act<<<(unsigned)CDIV(NB * 256, 256), 256, 0, stream>>>(
        hA, 256, bp.w2, bp.b2, command, hB, 256, NB, 256, 256, 1);
    branch_linear_act<<<1, 64, 0, stream>>>(
        hB, 256, bp.w3, bp.b3, command, dst, 3, NB, 256, 1, act);
  };
  run_branch(str, /*tanh*/3,    out + 0);  // steer   -> col 0
  run_branch(thr, /*sigmoid*/2, out + 1);  // throttle-> col 1
  run_branch(brk, /*sigmoid*/2, out + 2);  // brake   -> col 2
}
